// KWS12VerifierNet_36051955482629
// MI455X (gfx1250) — matrix operand order repaired and checked by reading, not compiled, not run on hardware
//
#include <hip/hip_runtime.h>
#include <math.h>

typedef __attribute__((ext_vector_type(16))) _Float16 v16h;
typedef __attribute__((ext_vector_type(8)))  float    v8f;
typedef __attribute__((ext_vector_type(4)))  int      v4i;
typedef _Float16 h16;

#define DEV __device__ __forceinline__

#define NMELS 80
#define TT    1000
#define BB    32
#define CC_   32
#define AD_   96
#define NH_   4
#define DH    24

#if __has_builtin(__builtin_amdgcn_global_load_async_to_lds_b128)
#define HAVE_ASYNC_LDS 1
#endif

DEV float gelu_f(float x) { return 0.5f * x * (1.0f + erff(x * 0.70710678f)); }

DEV v8f zero8() {
  v8f z;
#pragma unroll
  for (int i = 0; i < 8; ++i) z[i] = 0.f;
  return z;
}

DEV void wait_async_then_sync() {
#ifdef HAVE_ASYNC_LDS
#if __has_builtin(__builtin_amdgcn_s_wait_asynccnt)
  __builtin_amdgcn_s_wait_asynccnt(0);
#else
  asm volatile("s_wait_asynccnt 0" ::: "memory");
#endif
#endif
  __syncthreads();
}

// Cooperative copy of `bytes` (multiple of 16) from global f16 to LDS f16.
// Uses gfx1250 async global->LDS DMA when available.
DEV void stage_g2l(const h16* gsrc, h16* ldst, int bytes, int tid, int nthr) {
#ifdef HAVE_ASYNC_LDS
  typedef __attribute__((address_space(1))) v4i gv4;
  typedef __attribute__((address_space(3))) v4i lv4;
  for (int off = tid * 16; off < bytes; off += nthr * 16) {
    __builtin_amdgcn_global_load_async_to_lds_b128(
        (gv4*)(void*)((const char*)gsrc + off),
        (lv4*)(void*)((char*)ldst + off), 0, 0);
  }
#else
  int n = bytes >> 1;
  for (int i = tid; i < n; i += nthr) ldst[i] = gsrc[i];
#endif
}

// A fragment: 16x32 f16 matrix, row-major with row stride lda.
// ISA 7.12.2: lanes 0-15 -> M=lane, elems 0..7 = K0..7, elems 8..15 = K16..23;
//             lanes 16-31 -> M=lane-16, elems 0..7 = K8..15, elems 8..15 = K24..31.
DEV v16h load_A(const h16* Ap, int lda, int lane) {
  int m  = lane & 15;
  int kb = (lane & 16) ? 8 : 0;
  v16h a;
#pragma unroll
  for (int i = 0; i < 8; ++i) {
    a[i]     = Ap[m * lda + kb + i];
    a[8 + i] = Ap[m * lda + kb + 16 + i];
  }
  return a;
}

// B fragment: 32x16 f16 matrix (row = K, col = N), row stride ldb.
// lanes 0-15: N=lane, elems 0..7 = K0..7, elems 8..15 = K16..23;
// lanes 16-31: N=lane-16, elems 0..7 = K8..15, elems 8..15 = K24..31.
DEV v16h load_B(const h16* Bp, int ldb, int lane) {
  int n  = lane & 15;
  int kb = (lane & 16) ? 8 : 0;
  v16h b;
#pragma unroll
  for (int i = 0; i < 16; ++i) b[i] = Bp[(kb + (i & 7) + ((i & 8) << 1)) * ldb + n];
  return b;
}

DEV v8f wmma16(v16h a, v16h b, v8f c) {
  return __builtin_amdgcn_wmma_f32_16x16x32_f16(false, a, false, b, (short)0, c,
                                                false, false);
}

// ---------------------------------------------------------------- PCEN
__global__ void pcen_kernel(const float* feat, const float* logit_s,
                            const float* alpha, const float* delta,
                            const float* log_r, float* out) {
  int i = blockIdx.x * blockDim.x + threadIdx.x;
  if (i >= BB * NMELS) return;
  int mel = i % NMELS;
  const float* xr = feat + (long long)i * TT;
  float* orow = out + (long long)i * TT;
  float s = 1.0f / (1.0f + expf(-logit_s[mel]));
  float a = fminf(fmaxf(alpha[mel], 0.1f), 1.0f);
  float d = fmaxf(delta[mel], 0.1f);
  float r = fminf(fmaxf(expf(log_r[mel]), 0.05f), 1.5f);
  float dr = powf(d, r);
  float m = 0.f;
  for (int t = 0; t < TT; ++t) {
    float xv = fmaxf(xr[t], 0.f);
    m = (t == 0) ? xv : (1.f - s) * m + s * xv;
    orow[t] = powf(xv / powf(1e-6f + m, a) + d, r) - dr;
  }
}

// ----------------------------------------------- stem: 5x5 conv + BN + GELU
// out layout channels-last: x[((b*80+m)*1000+t)*32 + c]  (f16)
__global__ void stem_kernel(const float* pcen, const float* w, const float* g,
                            const float* bta, const float* mu, const float* var,
                            h16* x) {
  int i = blockIdx.x * blockDim.x + threadIdx.x;
  if (i >= BB * NMELS * TT * CC_) return;
  int c = i & 31;
  int p = i >> 5;
  int t = p % TT;
  int bm = p / TT;
  int m = bm % NMELS;
  int b = bm / NMELS;
  const float* pc = pcen + (long long)b * NMELS * TT;
  float acc = 0.f;
#pragma unroll
  for (int ky = 0; ky < 5; ++ky) {
    int mm = m + ky - 2;
    if (mm < 0 || mm >= NMELS) continue;
#pragma unroll
    for (int kx = 0; kx < 5; ++kx) {
      int tt = t + kx - 2;
      if (tt < 0 || tt >= TT) continue;
      acc += pc[mm * TT + tt] * w[c * 25 + ky * 5 + kx];
    }
  }
  float inv = g[c] * rsqrtf(var[c] + 1e-5f);
  x[i] = (h16)gelu_f(acc * inv + (bta[c] - mu[c] * inv));
}

// ----------------------------------------------- depthwise 3x3 + BN1 + GELU
__global__ void dw_kernel(const h16* x, const float* w, const float* g,
                          const float* bta, const float* mu, const float* var,
                          h16* h) {
  int i = blockIdx.x * blockDim.x + threadIdx.x;
  if (i >= BB * NMELS * TT * CC_) return;
  int c = i & 31;
  int p = i >> 5;
  int t = p % TT;
  int bm = p / TT;
  int m = bm % NMELS;
  int b = bm / NMELS;
  float acc = 0.f;
#pragma unroll
  for (int ky = 0; ky < 3; ++ky) {
    int mm = m + ky - 1;
    if (mm < 0 || mm >= NMELS) continue;
#pragma unroll
    for (int kx = 0; kx < 3; ++kx) {
      int tt = t + kx - 1;
      if (tt < 0 || tt >= TT) continue;
      acc += (float)x[((b * NMELS + mm) * TT + tt) * CC_ + c] * w[c * 9 + ky * 3 + kx];
    }
  }
  float inv = g[c] * rsqrtf(var[c] + 1e-5f);
  h[i] = (h16)gelu_f(acc * inv + (bta[c] - mu[c] * inv));
}

// ------------- fused pointwise chain: pw+bn2+gelu -> expand+gelu -> project
//               + residual + gelu.  One wave per 16 spatial positions.
//               Weights staged into LDS via async global->LDS DMA.
__global__ void chain_kernel(const h16* hbuf, h16* xbuf, const h16* pw_t,
                             const h16* ex_t, const h16* pr_t, const float* g2,
                             const float* b2, const float* m2, const float* v2) {
  __shared__ h16 Wl[5120];  // pw[0..1023] ex[1024..3071] pr[3072..5119]
  __shared__ h16 t2[4][16 * 32];
  __shared__ h16 t3[4][16 * 64];
  int tid = threadIdx.x;
  int wave = tid >> 5;
  int lane = tid & 31;
  stage_g2l(pw_t, Wl, 2048, tid, 128);
  stage_g2l(ex_t, Wl + 1024, 4096, tid, 128);
  stage_g2l(pr_t, Wl + 3072, 4096, tid, 128);
  wait_async_then_sync();

  int m0 = blockIdx.x * 64 + wave * 16;
  int nrow = lane & 15;
  int mbase = (lane & 16) ? 8 : 0;

  v16h aH = load_A(hbuf + (long long)m0 * 32, 32, lane);
  // pw (K=32 -> N=32) + bn2 + gelu
#pragma unroll
  for (int nt = 0; nt < 2; ++nt) {
    v8f acc = wmma16(aH, load_B(Wl + nt * 16, 32, lane), zero8());
    int c = nt * 16 + nrow;
    float inv = g2[c] * rsqrtf(v2[c] + 1e-5f);
    float sh = b2[c] - m2[c] * inv;
#pragma unroll
    for (int i = 0; i < 8; ++i)
      t2[wave][(mbase + i) * 32 + c] = (h16)gelu_f(acc[i] * inv + sh);
  }
  __syncthreads();
  v16h aT2 = load_A(t2[wave], 32, lane);
  // expand (K=32 -> N=64) + gelu
#pragma unroll
  for (int nt = 0; nt < 4; ++nt) {
    v8f acc = wmma16(aT2, load_B(Wl + 1024 + nt * 16, 64, lane), zero8());
    int c = nt * 16 + nrow;
#pragma unroll
    for (int i = 0; i < 8; ++i)
      t3[wave][(mbase + i) * 64 + c] = (h16)gelu_f(acc[i]);
  }
  __syncthreads();
  v16h aT3a = load_A(t3[wave], 64, lane);
  v16h aT3b = load_A(t3[wave] + 32, 64, lane);
  // project (K=64 -> N=32) + residual + gelu
#pragma unroll
  for (int nt = 0; nt < 2; ++nt) {
    v8f acc = wmma16(aT3a, load_B(Wl + 3072 + nt * 16, 32, lane), zero8());
    acc = wmma16(aT3b, load_B(Wl + 3072 + 32 * 32 + nt * 16, 32, lane), acc);
    int c = nt * 16 + nrow;
#pragma unroll
    for (int i = 0; i < 8; ++i) {
      long long row = m0 + mbase + i;
      float v = acc[i] + (float)xbuf[row * 32 + c];
      xbuf[row * 32 + c] = (h16)gelu_f(v);
    }
  }
}

// ------------------------------------------------ mean over mels, transpose
__global__ void meanpool_kernel(const h16* x, h16* pooled) {
  int i = blockIdx.x * blockDim.x + threadIdx.x;
  if (i >= BB * TT * CC_) return;
  int c = i & 31;
  int p = i >> 5;
  int t = p % TT;
  int b = p / TT;
  float s = 0.f;
  for (int m = 0; m < NMELS; ++m)
    s += (float)x[((b * NMELS + m) * TT + t) * CC_ + c];
  pooled[(b * TT + t) * CC_ + c] = (h16)(s * (1.f / NMELS));
}

// ------------------------------------------------- generic WMMA GEMM + bias
// A: [M][K] f16, Wt: [K][N] f16 (pre-transposed weights, staged to LDS),
// OUT: [M][N] f16
template <int K, int N, bool DO_GELU>
__global__ void gemm_kernel(const h16* A, const h16* Wt, const float* bias,
                            h16* OUT, int M) {
  __shared__ h16 Wl[K * N];
  int tid = threadIdx.x;
  stage_g2l(Wt, Wl, K * N * 2, tid, 256);
  wait_async_then_sync();

  int wave = tid >> 5;
  int lane = tid & 31;
  int m0 = (blockIdx.x * (blockDim.x >> 5) + wave) * 16;
  if (m0 >= M) return;
  int nrow = lane & 15;
  int mbase = (lane & 16) ? 8 : 0;
  v16h afr[K / 32];
#pragma unroll
  for (int kt = 0; kt < K / 32; ++kt)
    afr[kt] = load_A(A + (long long)m0 * K + kt * 32, K, lane);
#pragma unroll
  for (int nt = 0; nt < N / 16; ++nt) {
    v8f acc = zero8();
#pragma unroll
    for (int kt = 0; kt < K / 32; ++kt)
      acc = wmma16(afr[kt], load_B(Wl + kt * 32 * N + nt * 16, N, lane), acc);
    int n = nt * 16 + nrow;
    float bi = bias[n];
#pragma unroll
    for (int i = 0; i < 8; ++i) {
      float v = acc[i] + bi;
      if (DO_GELU) v = gelu_f(v);
      OUT[(long long)(m0 + mbase + i) * N + n] = (h16)v;
    }
  }
}

// ---------------------------------------- flash-style attention, 1 wave/blk
__global__ void attn_kernel(const h16* qkv, h16* outa) {
  __shared__ h16 Qt[16 * 32];
  __shared__ h16 Kt[32 * 32];  // [feature][key]
  __shared__ h16 Vt[32 * 32];  // [key][feature]
  __shared__ h16 Pt[16 * 32];
  int lane = threadIdx.x;
  int qt = blockIdx.x % 63;
  int bh = blockIdx.x / 63;
  int h = bh % NH_;
  int b = bh / NH_;
  const float scale = 0.20412414523193154f;  // 24^-0.5
  for (int idx = lane; idx < 16 * 32; idx += 32) {
    int r = idx >> 5, f = idx & 31;
    int t = qt * 16 + r;
    float v = 0.f;
    if (f < DH && t < TT) v = (float)qkv[(long long)(b * TT + t) * 288 + h * DH + f];
    Qt[idx] = (h16)v;
  }
  __syncthreads();
  v16h aQ = load_A(Qt, 32, lane);
  int nrow = lane & 15;
  int mbase = (lane & 16) ? 8 : 0;
  float rm[8], rl[8];
  v8f acc0 = zero8(), acc1 = zero8();
#pragma unroll
  for (int i = 0; i < 8; ++i) { rm[i] = -1e30f; rl[i] = 0.f; }

  for (int kb = 0; kb < 32; ++kb) {
    // prefetch next key block (gfx1250 global_prefetch_b8)
    if (kb + 1 < 32) {
      int pk = (kb + 1) * 32 + lane;
      pk = pk < TT ? pk : TT - 1;
      const h16* pfb = qkv + (long long)(b * TT + pk) * 288 + h * DH;
      __builtin_prefetch(pfb + 96, 0, 3);
      __builtin_prefetch(pfb + 192, 0, 3);
    }
    for (int idx = lane; idx < 1024; idx += 32) {
      int f = idx >> 5, key = idx & 31;
      int kg = kb * 32 + key;
      float kv = 0.f, vv = 0.f;
      if (f < DH && kg < TT) {
        long long base = (long long)(b * TT + kg) * 288 + h * DH + f;
        kv = (float)qkv[base + 96];
        vv = (float)qkv[base + 192];
      }
      Kt[f * 32 + key] = (h16)kv;
      Vt[key * 32 + f] = (h16)vv;
    }
    __syncthreads();
    v8f s0 = wmma16(aQ, load_B(Kt, 32, lane), zero8());
    v8f s1 = wmma16(aQ, load_B(Kt + 16, 32, lane), zero8());
    int k0 = kb * 32 + nrow, k1 = kb * 32 + 16 + nrow;
#pragma unroll
    for (int i = 0; i < 8; ++i) {
      s0[i] = (k0 < TT) ? s0[i] * scale : -1e30f;
      s1[i] = (k1 < TT) ? s1[i] * scale : -1e30f;
      float lm = fmaxf(s0[i], s1[i]);
#pragma unroll
      for (int off = 8; off >= 1; off >>= 1) lm = fmaxf(lm, __shfl_xor(lm, off, 32));
      float nm = fmaxf(rm[i], lm);
      float corr = expf(rm[i] - nm);
      float p0 = expf(s0[i] - nm), p1 = expf(s1[i] - nm);
      float ls = p0 + p1;
#pragma unroll
      for (int off = 8; off >= 1; off >>= 1) ls += __shfl_xor(ls, off, 32);
      rl[i] = rl[i] * corr + ls;
      rm[i] = nm;
      acc0[i] *= corr;
      acc1[i] *= corr;
      Pt[(mbase + i) * 32 + nrow] = (h16)p0;
      Pt[(mbase + i) * 32 + 16 + nrow] = (h16)p1;
    }
    __syncthreads();
    v16h aP = load_A(Pt, 32, lane);
    acc0 = wmma16(aP, load_B(Vt, 32, lane), acc0);
    acc1 = wmma16(aP, load_B(Vt + 16, 32, lane), acc1);
    __syncthreads();
  }
#pragma unroll
  for (int i = 0; i < 8; ++i) {
    int t = qt * 16 + mbase + i;
    if (t < TT) {
      float inv = 1.0f / rl[i];
      outa[(long long)(b * TT + t) * 96 + h * DH + nrow] = (h16)(acc0[i] * inv);
      if (16 + nrow < DH)
        outa[(long long)(b * TT + t) * 96 + h * DH + 16 + nrow] = (h16)(acc1[i] * inv);
    }
  }
}

// ------------------------------------------- residual add + LayerNorm (f32)
__global__ void addln_kernel(const h16* xproj, const h16* aproj, const float* g,
                             const float* bta, float* x2) {
  int r = blockIdx.x * blockDim.x + threadIdx.x;
  if (r >= BB * TT) return;
  float buf[96];
  float mu = 0.f;
  for (int j = 0; j < 96; ++j) {
    buf[j] = (float)xproj[(long long)r * 96 + j] + (float)aproj[(long long)r * 96 + j];
    mu += buf[j];
  }
  mu *= (1.f / 96.f);
  float var = 0.f;
  for (int j = 0; j < 96; ++j) { float d = buf[j] - mu; var += d * d; }
  var *= (1.f / 96.f);
  float inv = rsqrtf(var + 1e-5f);
  for (int j = 0; j < 96; ++j)
    x2[(long long)r * 96 + j] = (buf[j] - mu) * inv * g[j] + bta[j];
}

__global__ void pool2_kernel(const float* x2, float* pooled2) {
  int i = blockIdx.x * blockDim.x + threadIdx.x;
  if (i >= BB * 96) return;
  int b = i / 96, f = i % 96;
  float s = 0.f, mx = -1e30f;
  for (int t = 0; t < TT; ++t) {
    float v = x2[(long long)(b * TT + t) * 96 + f];
    s += v;
    mx = fmaxf(mx, v);
  }
  pooled2[b * 192 + f] = s * (1.f / TT);
  pooled2[b * 192 + 96 + f] = mx;
}

__global__ void emb_kernel(const float* pooled2, const float* w,
                           const float* bias, float* emb, float* dout) {
  int i = blockIdx.x * blockDim.x + threadIdx.x;
  if (i >= BB * 96) return;
  int b = i / 96, o = i % 96;
  float s = bias[o];
  for (int j = 0; j < 192; ++j) s += pooled2[b * 192 + j] * w[o * 192 + j];
  s = gelu_f(s);
  emb[b * 96 + o] = s;
  dout[BB * 12 + b * 96 + o] = s;
}

__global__ void logits_kernel(const float* emb, const float* w,
                              const float* bias, float* dout) {
  int i = blockIdx.x * blockDim.x + threadIdx.x;
  if (i >= BB * 12) return;
  int b = i / 12, c = i % 12;
  float s = bias[c];
  for (int j = 0; j < 96; ++j) s += emb[b * 96 + j] * w[c * 96 + j];
  dout[b * 12 + c] = s;
}

// weight [N][K] f32  ->  [K][N] f16
__global__ void wtT_kernel(const float* W, h16* Wt, int N, int K) {
  int i = blockIdx.x * blockDim.x + threadIdx.x;
  if (i >= N * K) return;
  int k = i / N, n = i % N;
  Wt[k * N + n] = (h16)W[n * K + k];
}

// ---------------------------------------------------------------------------
extern "C" void kernel_launch(void* const* d_in, const int* in_sizes, int n_in,
                              void* d_out, int out_size, void* d_ws,
                              size_t ws_size, hipStream_t stream) {
  (void)in_sizes; (void)n_in; (void)out_size; (void)ws_size;
  auto F = [&](int i) { return (const float*)d_in[i]; };

  char* ws = (char*)d_ws;
  const size_t OFF_X = 10240000;             // after pcen (f32 2.56M)
  const size_t OFF_H = OFF_X + 163840000;    // x f16 (81.92M elems)
  const size_t OFF_WT = OFF_H + 163840000;   // h f16
  float* pcen = (float*)ws;
  h16* x = (h16*)(ws + OFF_X);
  h16* h = (h16*)(ws + OFF_H);
  // post-conv buffers overlay the h region (h is dead after the last chain)
  h16* pooled = (h16*)(ws + OFF_H);                 // 2,048,000 B
  h16* xproj = (h16*)(ws + OFF_H + 2048000);        // 6,144,000 B
  h16* qkvb = (h16*)(ws + OFF_H + 8192000);         // 18,432,000 B
  h16* attnb = (h16*)(ws + OFF_H + 26624000);       // 6,144,000 B
  h16* aproj = (h16*)(ws + OFF_H + 32768000);       // 6,144,000 B
  float* x2 = (float*)(ws + OFF_H + 38912000);      // 12,288,000 B
  float* pool2 = (float*)(ws + OFF_H + 51200000);   // 24,576 B
  float* embbuf = (float*)(ws + OFF_H + 51230720);  // 12,288 B
  h16* wts = (h16*)(ws + OFF_WT);
  // f16 weight sublayout (element offsets)
  h16* pw_t[3] = {wts + 0, wts + 1024, wts + 2048};
  h16* ex_t[3] = {wts + 3072, wts + 5120, wts + 7168};
  h16* pr_t[3] = {wts + 9216, wts + 11264, wts + 13312};
  h16* proj_t = wts + 15360;    // [32][96]
  h16* qkvw_t = wts + 18432;    // [96][288]
  h16* outw_t = wts + 46080;    // [96][96]

  // ---- weight conversion/transposition
  for (int blk = 0; blk < 3; ++blk) {
    int base = 10 + 12 * blk;
    wtT_kernel<<<(32 * 32 + 255) / 256, 256, 0, stream>>>(F(base + 1), pw_t[blk], 32, 32);
    wtT_kernel<<<(64 * 32 + 255) / 256, 256, 0, stream>>>(F(base + 10), ex_t[blk], 64, 32);
    wtT_kernel<<<(32 * 64 + 255) / 256, 256, 0, stream>>>(F(base + 11), pr_t[blk], 32, 64);
  }
  wtT_kernel<<<(96 * 32 + 255) / 256, 256, 0, stream>>>(F(46), proj_t, 96, 32);
  wtT_kernel<<<(288 * 96 + 255) / 256, 256, 0, stream>>>(F(48), qkvw_t, 288, 96);
  wtT_kernel<<<(96 * 96 + 255) / 256, 256, 0, stream>>>(F(50), outw_t, 96, 96);

  // ---- PCEN
  pcen_kernel<<<(BB * NMELS + 127) / 128, 128, 0, stream>>>(F(0), F(1), F(2), F(3), F(4), pcen);

  // ---- stem
  const int NELT = BB * NMELS * TT * CC_;  // 81,920,000
  stem_kernel<<<NELT / 256, 256, 0, stream>>>(pcen, F(5), F(6), F(7), F(8), F(9), x);

  // ---- blocks
  for (int blk = 0; blk < 3; ++blk) {
    int base = 10 + 12 * blk;
    dw_kernel<<<NELT / 256, 256, 0, stream>>>(x, F(base + 0), F(base + 2), F(base + 3),
                                              F(base + 4), F(base + 5), h);
    chain_kernel<<<40000, 128, 0, stream>>>(h, x, pw_t[blk], ex_t[blk], pr_t[blk],
                                            F(base + 6), F(base + 7), F(base + 8),
                                            F(base + 9));
  }

  // ---- mean over mels -> (B*T, 32)
  meanpool_kernel<<<(BB * TT * CC_) / 256, 256, 0, stream>>>(x, pooled);

  const int M = BB * TT;  // 32000
  // ---- proj 32->96
  gemm_kernel<32, 96, false><<<M / 128, 256, 0, stream>>>(pooled, proj_t, F(47), xproj, M);
  // ---- qkv 96->288
  gemm_kernel<96, 288, false><<<M / 128, 256, 0, stream>>>(xproj, qkvw_t, F(49), qkvb, M);
  // ---- attention
  attn_kernel<<<BB * NH_ * 63, 32, 0, stream>>>(qkvb, attnb);
  // ---- out proj 96->96
  gemm_kernel<96, 96, false><<<M / 128, 256, 0, stream>>>(attnb, outw_t, F(51), aproj, M);
  // ---- residual + layernorm
  addln_kernel<<<(M + 127) / 128, 128, 0, stream>>>(xproj, aproj, F(52), F(53), x2);
  // ---- mean/max pool over time
  pool2_kernel<<<(BB * 96 + 127) / 128, 128, 0, stream>>>(x2, pool2);
  // ---- embedding MLP + logits
  emb_kernel<<<(BB * 96 + 127) / 128, 128, 0, stream>>>(pool2, F(54), F(55), embbuf,
                                                        (float*)d_out);
  logits_kernel<<<(BB * 12 + 127) / 128, 128, 0, stream>>>(embbuf, F(56), F(57),
                                                           (float*)d_out);
}
